// FRAP_8478265442456
// MI455X (gfx1250) — matrix operand order repaired and checked by reading, not compiled, not run on hardware
//
#include <hip/hip_runtime.h>

typedef __attribute__((ext_vector_type(16))) _Float16 v16h;
typedef __attribute__((ext_vector_type(8)))  float    v8f;

#define NPHASE 8
#define NMOV   12
#define LEU    16
#define AG_PER_BLOCK 128
#define THREADS 256

union AFrag { v16h v; uint4 u[2]; };

__device__ __forceinline__ float sigmoidf_(float x) {
    return 1.0f / (1.0f + __expf(-x));
}

__global__ __launch_bounds__(THREADS)
void frap_wmma_kernel(const float* __restrict__ states,
                      const int*   __restrict__ comp_mask,
                      const int*   __restrict__ phase_pairs,
                      const float* __restrict__ p_w,
                      const float* __restrict__ d_w,
                      const float* __restrict__ d_b,
                      const float* __restrict__ le_w,
                      const float* __restrict__ le_b,
                      const float* __restrict__ lc_w,
                      const float* __restrict__ lc_b,
                      const float* __restrict__ rel_w,
                      const float* __restrict__ rc_w,
                      const float* __restrict__ rc_b,
                      const float* __restrict__ h_w,
                      const float* __restrict__ h_b,
                      const float* __restrict__ bm_w,
                      const float* __restrict__ bm_b,
                      float* __restrict__ out)
{
    // pairs features per agent: [128 agents][8 phases][16] f16  (32 KB)
    __shared__ __align__(16) _Float16 sPairs[AG_PER_BLOCK * NPHASE * LEU];
    // batch-independent relation gate: [56][20] f32 (4.5 KB)
    __shared__ __align__(16) float sRel[NPHASE * (NPHASE - 1) * 20];
    // per-wave transpose staging tile: [wave][16 rows][32 cols] f16 (8 KB)
    __shared__ __align__(16) _Float16 sT[8][16][32];

    const int tid = threadIdx.x;

    // ---------------- Stage 0: rel = relu(rc(relu(rel_w[comp_mask]))) ----------------
    for (int idx = tid; idx < NPHASE * (NPHASE - 1) * 20; idx += THREADS) {
        int pq = idx / 20, o = idx % 20;
        int pp = pq / 7, qq = pq % 7;
        int cm = comp_mask[pp * 7 + qq];
        float a = rc_b[o];
#pragma unroll
        for (int c = 0; c < 4; ++c)
            a += fmaxf(rel_w[cm * 4 + c], 0.f) * rc_w[o * 4 + c];
        sRel[idx] = fmaxf(a, 0.f);
    }

    // ---------------- Stage 1: per-agent front-end -> pairs[8][16] ----------------
    if (tid < AG_PER_BLOCK) {
        const int agent = blockIdx.x * AG_PER_BLOCK + tid;
        const float* st = states + agent * 13;
        const int act = (int)st[0];
        const int pr0 = phase_pairs[act * 2 + 0];
        const int pr1 = phase_pairs[act * 2 + 1];

        float sp0[4], sp1[4], leb[LEU];
#pragma unroll
        for (int k = 0; k < 4; ++k) { sp0[k] = sigmoidf_(p_w[k]); sp1[k] = sigmoidf_(p_w[4 + k]); }
#pragma unroll
        for (int j = 0; j < LEU; ++j) leb[j] = le_b[j];

        for (int p = 0; p < NPHASE; ++p) {
            float pv[LEU];
#pragma unroll
            for (int j = 0; j < LEU; ++j) pv[j] = 0.f;

#pragma unroll
            for (int s = 0; s < 2; ++s) {
                const int m = phase_pairs[p * 2 + s];
                const int e = (m == pr0) || (m == pr1);
                float emb[8];
#pragma unroll
                for (int k = 0; k < 4; ++k) emb[k] = e ? sp1[k] : sp0[k];
                const float dm = st[1 + m];
#pragma unroll
                for (int k = 0; k < 4; ++k) emb[4 + k] = sigmoidf_(dm * d_w[k] + d_b[k]);

                float lm[LEU];
#pragma unroll
                for (int j = 0; j < LEU; ++j) lm[j] = leb[j];
#pragma unroll
                for (int k = 0; k < 8; ++k) {
                    const float ek = emb[k];
#pragma unroll
                    for (int j = 0; j < LEU; ++j) lm[j] += ek * le_w[k * LEU + j];
                }
#pragma unroll
                for (int j = 0; j < LEU; ++j) pv[j] += fmaxf(lm[j], 0.f);
            }
#pragma unroll
            for (int j = 0; j < LEU; ++j)
                sPairs[(tid * NPHASE + p) * LEU + j] = (_Float16)pv[j];
        }
    }

    // ---------------- Stage 2 setup: per-wave constant WMMA B fragments ----------------
    const int lane = tid & 31;
    const int wave = tid >> 5;
    const int n    = lane & 15;     // output-channel column (D layout: lane -> N)
    const int hi   = lane >> 4;     // lane-half selector
    const int n1   = 16 + n;        // second N tile channel
    const bool v1  = (n1 < 20);

    v16h Blc0, Blc1, Bh0, Bh1;
#pragma unroll
    for (int e = 0; e < 16; ++e) {
        const int k = (e < 8) ? (8 * hi + e) : (16 + 8 * hi + (e - 8));  // K index this half-element holds (B layout)
        Blc0[e] = (_Float16)lc_w[n * 32 + k];                                   // n < 16 < 20
        Blc1[e] = v1 ? (_Float16)lc_w[n1 * 32 + k] : (_Float16)0.f;
        Bh0[e]  = (k < 20) ? (_Float16)h_w[n * 20 + k] : (_Float16)0.f;         // K pad 20->32
        Bh1[e]  = (v1 && k < 20) ? (_Float16)h_w[n1 * 20 + k] : (_Float16)0.f;
    }
    const float lcb0 = lc_b[n];
    const float lcb1 = v1 ? lc_b[n1] : 0.f;
    const float hb0  = h_b[n];
    const float hb1  = v1 ? h_b[n1] : 0.f;
    const float bmw0 = bm_w[n];
    const float bmw1 = v1 ? bm_w[n1] : 0.f;
    const float bmb  = bm_b[0];

    __syncthreads();

    const int wAgent0 = wave * 16;  // this wave's 16-agent M-tile (local)

    for (int p = 0; p < NPHASE; ++p) {
        float acc[8];
#pragma unroll
        for (int r = 0; r < 8; ++r) acc[r] = 0.f;

        for (int q = 0; q < NPHASE - 1; ++q) {
            const int j  = (q < p) ? q : q + 1;
            const int pq = p * (NPHASE - 1) + q;

            // A fragment (16x32 f16): lane holds row m=n; halves 0..7 = pairs[p][8*hi..],
            // halves 8..15 = pairs[j][8*hi..]  (matches 16-bit A layout)
            AFrag A;
            A.u[0] = *(const uint4*)&sPairs[((wAgent0 + n) * NPHASE + p) * LEU + 8 * hi];
            A.u[1] = *(const uint4*)&sPairs[((wAgent0 + n) * NPHASE + j) * LEU + 8 * hi];

            v8f c0 = {}; v8f c1 = {};
            c0 = __builtin_amdgcn_wmma_f32_16x16x32_f16(false, A.v, false, Blc0,
                                                        (short)0, c0, false, false);
            c1 = __builtin_amdgcn_wmma_f32_16x16x32_f16(false, A.v, false, Blc1,
                                                        (short)0, c1, false, false);

            const float r0 = sRel[pq * 20 + n];
            const float r1 = v1 ? sRel[pq * 20 + n1] : 0.f;

            // relu(+bias) * rel, stage into LDS to transpose D-layout -> A-layout
#pragma unroll
            for (int r = 0; r < 8; ++r) {
                const float t0 = fmaxf(c0[r] + lcb0, 0.f) * r0;
                const float t1 = fmaxf(c1[r] + lcb1, 0.f) * r1;  // 0 for padded cols
                sT[wave][r + 8 * hi][n]      = (_Float16)t0;
                sT[wave][r + 8 * hi][16 + n] = (_Float16)t1;
            }
            asm volatile("s_wait_dscnt 0" ::: "memory");

            AFrag Ah;
            Ah.u[0] = *(const uint4*)&sT[wave][n][8 * hi];
            Ah.u[1] = *(const uint4*)&sT[wave][n][16 + 8 * hi];

            v8f d0 = {}; v8f d1 = {};
            d0 = __builtin_amdgcn_wmma_f32_16x16x32_f16(false, Ah.v, false, Bh0,
                                                        (short)0, d0, false, false);
            d1 = __builtin_amdgcn_wmma_f32_16x16x32_f16(false, Ah.v, false, Bh1,
                                                        (short)0, d1, false, false);

            // bm dot folded per-lane; reduce across N later
#pragma unroll
            for (int r = 0; r < 8; ++r) {
                const float u0 = fmaxf(d0[r] + hb0, 0.f);
                const float u1 = fmaxf(d1[r] + hb1, 0.f);
                acc[r] += u0 * bmw0 + u1 * bmw1;
            }
        }

        // butterfly-sum over the 16 lanes of each half (lanes share the same 8 rows)
#pragma unroll
        for (int r = 0; r < 8; ++r) {
            float v = acc[r];
            v += __shfl_xor(v, 1);
            v += __shfl_xor(v, 2);
            v += __shfl_xor(v, 4);
            v += __shfl_xor(v, 8);
            acc[r] = v + 7.f * bmb;   // bm_b added once per q (7 q's)
        }

        if (n == 0) {
            const int aBase = blockIdx.x * AG_PER_BLOCK + wAgent0 + 8 * hi;
#pragma unroll
            for (int r = 0; r < 8; ++r)
                out[(aBase + r) * NPHASE + p] = acc[r];
        }
    }
}

extern "C" void kernel_launch(void* const* d_in, const int* in_sizes, int n_in,
                              void* d_out, int out_size, void* d_ws, size_t ws_size,
                              hipStream_t stream) {
    const float* states      = (const float*)d_in[0];
    const int*   comp_mask   = (const int*)  d_in[1];
    const int*   phase_pairs = (const int*)  d_in[2];
    const float* p_w  = (const float*)d_in[3];
    const float* d_w  = (const float*)d_in[4];
    const float* d_b  = (const float*)d_in[5];
    const float* le_w = (const float*)d_in[6];
    const float* le_b = (const float*)d_in[7];
    const float* lc_w = (const float*)d_in[8];
    const float* lc_b = (const float*)d_in[9];
    const float* rel_w = (const float*)d_in[10];
    const float* rc_w  = (const float*)d_in[11];
    const float* rc_b  = (const float*)d_in[12];
    const float* h_w   = (const float*)d_in[13];
    const float* h_b   = (const float*)d_in[14];
    const float* bm_w  = (const float*)d_in[15];
    const float* bm_b  = (const float*)d_in[16];
    float* out = (float*)d_out;

    const int B = in_sizes[0] / 13;
    const int blocks = B / AG_PER_BLOCK;

    frap_wmma_kernel<<<blocks, THREADS, 0, stream>>>(
        states, comp_mask, phase_pairs, p_w, d_w, d_b, le_w, le_b,
        lc_w, lc_b, rel_w, rc_w, rc_b, h_w, h_b, bm_w, bm_b, out);
}
